// Model_60765197304065
// MI455X (gfx1250) — matrix operand order repaired and checked by reading, not compiled, not run on hardware
//
#include <hip/hip_runtime.h>

// CDNA5 / gfx1250 fused MLP + expert head + MSE loss.
// B=131072, D_IN=1024, E1=512, E2=128, N_ASSAY=4096.
// d_out layout: [0] = loss, [1..B] = affinity.

typedef __attribute__((ext_vector_type(16))) __bf16 v16bf;
typedef __attribute__((ext_vector_type(2)))  __bf16 v2bf;
typedef __attribute__((ext_vector_type(8)))  float  v8f;

#define B_TOT 131072
#define D_IN  1024
#define E1    512
#define E2    128
#define TM    128            // rows per workgroup
#define KC    64             // K-chunk staged in LDS
#define NBLK  (B_TOT / TM)   // 1024

// Padded LDS pitches (16B-multiple row strides for ds_load_b128 alignment,
// non-power-of-two bank strides to spread fragment loads across banks).
#define SAP 72    // sA pitch, bf16  (144B)
#define SWP 72    // sW1 pitch, bf16
#define SHP 520   // sH pitch, bf16  (1040B)
#define SCP 132   // sCA pitch, f32  (528B)

// LDS carve (220 KB total):
//   [0,     18432)  : sA  bf16[128][SAP]  (phase1 A-tile / phase2 W2t / phase3 sred)
//   [18432, 92160)  : sW1 bf16[512][SWP]  (phase1) | sCA f32[128][SCP] (phase3)
//   [92160, 225280) : sH  bf16[128][SHP]
#define OFF_B 18432
#define OFF_H 92160
#define SMEM_BYTES 225280

// Pack two f32 -> packed bf16 pair; vector-init form selects v_cvt_pk_bf16_f32.
__device__ __forceinline__ unsigned pk2(float a, float b) {
    v2bf p = { (__bf16)a, (__bf16)b };
    return __builtin_bit_cast(unsigned, p);
}

__device__ __forceinline__ v16bf load_frag32B(const __bf16* p0, const __bf16* p1) {
    v16bf f;
    ((uint4*)&f)[0] = *(const uint4*)p0;
    ((uint4*)&f)[1] = *(const uint4*)p1;
    return f;
}

__global__ __launch_bounds__(256, 1)
void fused_mlp_kernel(const float* __restrict__ comp,
                      const int*   __restrict__ assay,
                      const float* __restrict__ label,
                      const float* __restrict__ W1,
                      const float* __restrict__ b1,
                      const float* __restrict__ W2,
                      const float* __restrict__ b2,
                      const float* __restrict__ Wout,
                      const float* __restrict__ bout,
                      float* __restrict__ out,
                      float* __restrict__ partial) {
    __shared__ __align__(16) unsigned char smem[SMEM_BYTES];
    __bf16* sA  = (__bf16*)smem;                 // [128][SAP]
    __bf16* sW1 = (__bf16*)(smem + OFF_B);       // [512][SWP]
    float*  sCA = (float*) (smem + OFF_B);       // [128][SCP] (phase3)
    __bf16* sH  = (__bf16*)(smem + OFF_H);       // [128][SHP]

    const int tid  = threadIdx.x;
    const int wave = tid >> 5;
    const int lane = tid & 31;
    const int m    = lane & 15;   // row (A) or column (B/C) within 16x16 tile
    const int hi   = lane >> 4;   // lane-half selector per ISA wave32 layouts
    const size_t rowBase = (size_t)blockIdx.x * TM;

    // ---------------- Phase 1: h = relu(comp @ W1 + b1); 16x512 stripe per wave
    v8f acc[32];
    #pragma unroll
    for (int t = 0; t < 32; ++t) {
        v8f z = {0.f, 0.f, 0.f, 0.f, 0.f, 0.f, 0.f, 0.f};
        acc[t] = z;
    }

    for (int kc = 0; kc < D_IN; kc += KC) {
        // Stage comp[128][KC] -> bf16 sA. Batch all 8 b128 loads, then pack+store.
        {
            const float* src = comp + rowBase * D_IN + kc;
            if (kc + KC < D_IN)   // global_prefetch_b8 toward next chunk
                __builtin_prefetch(src + KC + (size_t)(tid >> 1) * D_IN, 0, 0);
            float4 v[8];
            #pragma unroll
            for (int i = 0; i < 8; ++i) {
                int l = tid + 256 * i;
                v[i] = *(const float4*)(src + (size_t)(l >> 4) * D_IN + ((l & 15) << 2));
            }
            #pragma unroll
            for (int i = 0; i < 8; ++i) {
                int l = tid + 256 * i;
                uint2 p;
                p.x = pk2(v[i].x, v[i].y);
                p.y = pk2(v[i].z, v[i].w);
                *(uint2*)(sA + (l >> 4) * SAP + ((l & 15) << 2)) = p;
            }
        }
        // Stage W1[kc:kc+KC][0:512] -> sW1[n][k]: coalesced b32 reads along N
        // (W1 is L2-resident), 4 contiguous bf16 per 8B store.
        {
            #pragma unroll
            for (int i = 0; i < 32; ++i) {
                int l  = tid + 256 * i;
                int n  = l & (E1 - 1);
                int kg = l >> 9;             // 0..15, group of 4 k's
                const float* p = W1 + (size_t)(kc + kg * 4) * E1 + n;
                float a0 = p[0], a1 = p[E1], a2 = p[2 * E1], a3 = p[3 * E1];
                uint2 pk;
                pk.x = pk2(a0, a1);
                pk.y = pk2(a2, a3);
                *(uint2*)(sW1 + n * SWP + kg * 4) = pk;
            }
        }
        __syncthreads();

        #pragma unroll
        for (int ks = 0; ks < 2; ++ks) {
            const int kb = ks * 32;
            const __bf16* ap = sA + (wave * 16 + m) * SAP + kb + hi * 16;
            v16bf afrag = load_frag32B(ap, ap + 8);
            const __bf16* bbase = sW1 + m * SWP + kb + hi * 16;
            // Depth-2 software pipeline on B fragments: waits relax to dscnt<=4.
            v16bf b0 = load_frag32B(bbase, bbase + 8);
            v16bf b1f = load_frag32B(bbase + 16 * SWP, bbase + 16 * SWP + 8);
            #pragma unroll
            for (int t = 0; t < 32; ++t) {
                v16bf bn;
                if (t < 30) {
                    const __bf16* bp = bbase + (t + 2) * 16 * SWP;
                    bn = load_frag32B(bp, bp + 8);
                }
                acc[t] = __builtin_amdgcn_wmma_f32_16x16x32_bf16(
                    false, afrag, false, b0, (short)0, acc[t], false, false);
                if (t < 31) b0 = b1f;
                if (t < 30) b1f = bn;
            }
        }
        __syncthreads();
    }

    // Epilogue 1: bias + relu -> sH (bf16)
    #pragma unroll
    for (int t = 0; t < 32; ++t) {
        int n = t * 16 + m;
        float bias = b1[n];
        #pragma unroll
        for (int r = 0; r < 8; ++r) {
            float v = acc[t][r] + bias;
            v = v > 0.f ? v : 0.f;
            sH[(wave * 16 + hi * 8 + r) * SHP + n] = (__bf16)v;
        }
    }
    __syncthreads();

    // ---------------- Phase 2: ca = relu(h @ W2 + b2); 16x128 stripe per wave
    v8f acc2[8];
    #pragma unroll
    for (int t = 0; t < 8; ++t) {
        v8f z = {0.f, 0.f, 0.f, 0.f, 0.f, 0.f, 0.f, 0.f};
        acc2[t] = z;
    }

    for (int kc = 0; kc < E1; kc += KC) {
        // Stage W2[kc:kc+KC][0:128] -> sA[n][k] (same N-major scheme)
        #pragma unroll
        for (int i = 0; i < 8; ++i) {
            int l  = tid + 256 * i;
            int n  = l & (E2 - 1);
            int kg = l >> 7;                 // 0..15
            const float* p = W2 + (size_t)(kc + kg * 4) * E2 + n;
            float a0 = p[0], a1 = p[E2], a2 = p[2 * E2], a3 = p[3 * E2];
            uint2 pk;
            pk.x = pk2(a0, a1);
            pk.y = pk2(a2, a3);
            *(uint2*)(sA + n * SAP + kg * 4) = pk;
        }
        __syncthreads();

        #pragma unroll
        for (int ks = 0; ks < 2; ++ks) {
            const int kb = ks * 32;
            const __bf16* ap = sH + (wave * 16 + m) * SHP + kc + kb + hi * 16;
            v16bf afrag = load_frag32B(ap, ap + 8);
            const __bf16* bbase = sA + m * SAP + kb + hi * 16;
            v16bf b0 = load_frag32B(bbase, bbase + 8);
            v16bf b1f = load_frag32B(bbase + 16 * SAP, bbase + 16 * SAP + 8);
            #pragma unroll
            for (int t = 0; t < 8; ++t) {
                v16bf bn;
                if (t < 6) {
                    const __bf16* bp = bbase + (t + 2) * 16 * SAP;
                    bn = load_frag32B(bp, bp + 8);
                }
                acc2[t] = __builtin_amdgcn_wmma_f32_16x16x32_bf16(
                    false, afrag, false, b0, (short)0, acc2[t], false, false);
                if (t < 7) b0 = b1f;
                if (t < 6) b1f = bn;
            }
        }
        __syncthreads();
    }

    // Epilogue 2: bias + relu -> sCA (f32; overlays sW1 region, safe post-barrier)
    #pragma unroll
    for (int t = 0; t < 8; ++t) {
        int n = t * 16 + m;
        float bias = b2[n];
        #pragma unroll
        for (int r = 0; r < 8; ++r) {
            float v = acc2[t][r] + bias;
            v = v > 0.f ? v : 0.f;
            sCA[(wave * 16 + hi * 8 + r) * SCP + n] = v;
        }
    }
    __syncthreads();

    // ---------------- Phase 3: gathered expert dot + squared error
    float* sred = (float*)smem;   // reuse region A as reduction scratch
    float sq = 0.f;
    if (tid < TM) {
        size_t gr = rowBase + tid;
        int aid = assay[gr];
        const float* wrow = Wout + (size_t)aid * E2;
        float dot = 0.f;
        #pragma unroll
        for (int e = 0; e < E2; e += 4) {
            float4 wv = *(const float4*)(wrow + e);
            const float* c = sCA + tid * SCP + e;
            dot += c[0] * wv.x + c[1] * wv.y + c[2] * wv.z + c[3] * wv.w;
        }
        float aff = dot + bout[aid];
        out[1 + gr] = aff;
        float d = aff - label[gr];
        sq = d * d;
    }
    sred[tid] = sq;
    #pragma unroll
    for (int s = 128; s >= 1; s >>= 1) {
        __syncthreads();
        if (tid < s) sred[tid] += sred[tid + s];
    }
    if (tid == 0) partial[blockIdx.x] = sred[0];
}

__global__ __launch_bounds__(256)
void loss_reduce_kernel(const float* __restrict__ partial, float* __restrict__ out) {
    __shared__ float s[256];
    float v = 0.f;
    for (int i = threadIdx.x; i < NBLK; i += 256) v += partial[i];
    s[threadIdx.x] = v;
    #pragma unroll
    for (int st = 128; st >= 1; st >>= 1) {
        __syncthreads();
        if (threadIdx.x < st) s[threadIdx.x] += s[threadIdx.x + st];
    }
    if (threadIdx.x == 0) out[0] = s[0] / (float)B_TOT;
}

extern "C" void kernel_launch(void* const* d_in, const int* in_sizes, int n_in,
                              void* d_out, int out_size, void* d_ws, size_t ws_size,
                              hipStream_t stream) {
    const float* comp  = (const float*)d_in[0];
    const int*   assay = (const int*)  d_in[1];
    const float* label = (const float*)d_in[2];
    const float* W1    = (const float*)d_in[3];
    const float* b1    = (const float*)d_in[4];
    const float* W2    = (const float*)d_in[5];
    const float* b2    = (const float*)d_in[6];
    const float* Wout  = (const float*)d_in[7];
    const float* bout  = (const float*)d_in[8];
    float* out     = (float*)d_out;
    float* partial = (float*)d_ws;   // NBLK floats

    hipLaunchKernelGGL(fused_mlp_kernel, dim3(NBLK), dim3(256), 0, stream,
                       comp, assay, label, W1, b1, W2, b2, Wout, bout, out, partial);
    hipLaunchKernelGGL(loss_reduce_kernel, dim3(1), dim3(256), 0, stream, partial, out);
}
